// NTXent_12652973654546
// MI455X (gfx1250) — matrix operand order repaired and checked by reading, not compiled, not run on hardware
//
#include <hip/hip_runtime.h>
#include <hip/hip_bf16.h>

typedef __attribute__((ext_vector_type(16))) _Float16 v16h;
typedef __attribute__((ext_vector_type(8)))  _Float16 v8h;
typedef __attribute__((ext_vector_type(8)))  float    v8f;

#define DDIM 128
#define TEMP_INV 2.0f                      // 1 / 0.5
#define EXP_SCALE 2.8853900817779268f      // TEMP_INV * log2(e)
#define LN2 0.6931471805599453f

// ---------------------------------------------------------------------------
// Kernel 1: row-wise L2 normalize (f32), emit f16 zn, store clamped norms,
// the self-similarity of the f16-rounded row (for diagonal removal), and
// zero the rowsum accumulator (ws is NOT re-zeroed between replays).
// One 128-thread block per row of the concatenated [2B, 128] matrix.
// ---------------------------------------------------------------------------
__global__ void ntx_normalize(const float* __restrict__ z_i,
                              const float* __restrict__ z_j,
                              _Float16* __restrict__ zn,
                              float* __restrict__ norms,
                              float* __restrict__ selfsim,
                              float* __restrict__ rowsum,
                              int B) {
  int row = blockIdx.x;
  int t   = threadIdx.x;                       // 0..127
  float v = (row < B) ? z_i[(size_t)row * DDIM + t]
                      : z_j[(size_t)(row - B) * DDIM + t];
  float sq = v * v;
  for (int m = 16; m; m >>= 1) sq += __shfl_xor(sq, m, 32);
  __shared__ float part[4];
  if ((t & 31) == 0) part[t >> 5] = sq;
  __syncthreads();
  float total = part[0] + part[1] + part[2] + part[3];
  float cl = fmaxf(sqrtf(total), 1e-8f);       // torch clamps norm at eps
  _Float16 h = (_Float16)(v / cl);
  zn[(size_t)row * DDIM + t] = h;

  // self-similarity of the f16-rounded row == what the WMMA diagonal computes
  float hf = (float)h;
  float sq2 = hf * hf;
  for (int m = 16; m; m >>= 1) sq2 += __shfl_xor(sq2, m, 32);
  __shared__ float part2[4];
  if ((t & 31) == 0) part2[t >> 5] = sq2;
  __syncthreads();
  if (t == 0) {
    norms[row]   = cl;
    selfsim[row] = part2[0] + part2[1] + part2[2] + part2[3];
    rowsum[row]  = 0.0f;
  }
}

// ---------------------------------------------------------------------------
// Kernel 2: positive-pair cosine sims in full f32
// ---------------------------------------------------------------------------
__global__ void ntx_pos(const float* __restrict__ z_i,
                        const float* __restrict__ z_j,
                        const float* __restrict__ norms,
                        float* __restrict__ pos,
                        int B) {
  int b = blockIdx.x;
  int t = threadIdx.x;                         // 0..127
  float p = z_i[(size_t)b * DDIM + t] * z_j[(size_t)b * DDIM + t];
  for (int m = 16; m; m >>= 1) p += __shfl_xor(p, m, 32);
  __shared__ float part[4];
  if ((t & 31) == 0) part[t >> 5] = p;
  __syncthreads();
  if (t == 0)
    pos[b] = (part[0] + part[1] + part[2] + part[3]) / (norms[b] * norms[b + B]);
}

// ---------------------------------------------------------------------------
// Kernel 3: fused  rowsum[i] += sum_j exp(sim[i][j]/T)  via f16 WMMA.
// (Diagonal j==i is INCLUDED here and subtracted in the finalize kernel, so
// the hot loop is branch-free: load / 4x WMMA / 8x exp / accumulate.)
// Block = 8 waves; wave w owns rows [blockIdx.x*128 + w*16, +16).
// blockIdx.y selects a chunk of 16-wide column tiles.
// Double-buffered B fragments; the unrolled-by-2 body is a SINGLE basic
// block (tile count per chunk is even; tail prefetch wraps to jStart) so the
// scheduler pipelines loads symmetrically across both WMMA groups.
// ---------------------------------------------------------------------------
__global__ __launch_bounds__(256) void ntx_simsum(const _Float16* __restrict__ zn,
                                                  float* __restrict__ rowsum,
                                                  int colTilesPerChunk) {
  const int lane   = threadIdx.x & 31;
  const int wave   = threadIdx.x >> 5;
  const int lane16 = lane & 15;
  const int hlf    = lane >> 4;                // 0 or 1
  const int rowBase = blockIdx.x * 128 + wave * 16;

  // ---- A fragments (loop-invariant, contiguous K = s*32 + hlf*16 + i, matches B) ----
  const _Float16* arow = zn + (size_t)(rowBase + lane16) * DDIM;
  v16h a[4];
#pragma unroll
  for (int s = 0; s < 4; ++s) {
    a[s] = *(const v16h*)(arow + s * 32 + hlf * 16);
  }

  float acc[8];
#pragma unroll
  for (int i = 0; i < 8; ++i) acc[i] = 0.0f;

  const int jStart = blockIdx.y * colTilesPerChunk;   // colTilesPerChunk is even
  const int jEnd   = jStart + colTilesPerChunk;

  // B fragment base for column n = j*16 + lane16, K = s*32 + hlf*16 + i
  const _Float16* bbase = zn + (size_t)lane16 * DDIM + hlf * 16;

  auto loadB = [&](int j, v16h* b) {
    const _Float16* p = bbase + (size_t)j * (16 * DDIM);
#pragma unroll
    for (int s = 0; s < 4; ++s) b[s] = *(const v16h*)(p + s * 32);
  };

  auto compute = [&](const v16h* b) {
    v8f c = {};
#pragma unroll
    for (int s = 0; s < 4; ++s) {
      c = __builtin_amdgcn_wmma_f32_16x16x32_f16(
          /*neg_a=*/false, a[s], /*neg_b=*/false, b[s],
          /*c_mod=*/(short)0, c, /*reuse_a=*/false, /*reuse_b=*/false);
    }
#pragma unroll
    for (int v = 0; v < 8; ++v)
      acc[v] += __builtin_amdgcn_exp2f(c[v] * EXP_SCALE);
  };

  // ---- software pipeline, unrolled by 2, no control flow in the body ----
  v16h b0[4], b1[4];
  loadB(jStart, b0);
  for (int j = jStart; j < jEnd; j += 2) {
    loadB(j + 1, b1);                            // always valid: even count
    compute(b0);
    int j2 = (j + 2 < jEnd) ? (j + 2) : jStart;  // scalar select; wrap = harmless
    loadB(j2, b0);
    compute(b1);
  }

  // Reduce across the 16 lanes holding one row (xor 1,2,4,8 stays in each half)
#pragma unroll
  for (int v = 0; v < 8; ++v) {
    float s = acc[v];
    s += __shfl_xor(s, 1, 32);
    s += __shfl_xor(s, 2, 32);
    s += __shfl_xor(s, 4, 32);
    s += __shfl_xor(s, 8, 32);
    if (lane16 == 0) atomicAdd(rowsum + rowBase + hlf * 8 + v, s);
  }
}

// ---------------------------------------------------------------------------
// Kernel 4: loss = mean_i( log(rowsum_i - exp(selfsim_i/T)) - pos_{i mod B}/T )
// ---------------------------------------------------------------------------
__global__ void ntx_finalize(const float* __restrict__ rowsum,
                             const float* __restrict__ selfsim,
                             const float* __restrict__ pos,
                             float* __restrict__ out, int B) {
  const int N = 2 * B;
  int t = threadIdx.x;                         // 256 threads, single block
  float local = 0.0f;
  for (int i = t; i < N; i += 256) {
    int pb = (i < B) ? i : i - B;
    float den = rowsum[i] - __builtin_amdgcn_exp2f(selfsim[i] * EXP_SCALE);
    local += __builtin_amdgcn_logf(den) * LN2 - pos[pb] * TEMP_INV;
  }
  for (int m = 16; m; m >>= 1) local += __shfl_xor(local, m, 32);
  __shared__ float part[8];
  if ((t & 31) == 0) part[t >> 5] = local;
  __syncthreads();
  if (t == 0) {
    float tot = 0.0f;
#pragma unroll
    for (int w = 0; w < 8; ++w) tot += part[w];
    out[0] = tot / (float)N;
  }
}

extern "C" void kernel_launch(void* const* d_in, const int* in_sizes, int n_in,
                              void* d_out, int out_size, void* d_ws, size_t ws_size,
                              hipStream_t stream) {
  const float* z_i = (const float*)d_in[0];
  const float* z_j = (const float*)d_in[1];
  const int B = in_sizes[0] / DDIM;            // 8192
  const int N = 2 * B;                         // 16384

  // workspace layout
  char* ws = (char*)d_ws;
  _Float16* zn = (_Float16*)ws;
  size_t off = (size_t)N * DDIM * sizeof(_Float16);   // 4 MB
  float* rowsum  = (float*)(ws + off); off += (size_t)N * sizeof(float);
  float* norms   = (float*)(ws + off); off += (size_t)N * sizeof(float);
  float* selfsim = (float*)(ws + off); off += (size_t)N * sizeof(float);
  float* pos     = (float*)(ws + off);

  ntx_normalize<<<N, DDIM, 0, stream>>>(z_i, z_j, zn, norms, selfsim, rowsum, B);
  ntx_pos<<<B, DDIM, 0, stream>>>(z_i, z_j, norms, pos, B);

  const int rowBlocks = N / 128;               // 128
  const int COLCHUNKS = 8;
  const int colTilesPerChunk = (N / 16) / COLCHUNKS;   // 128 (even, required)
  dim3 grid(rowBlocks, COLCHUNKS);
  ntx_simsum<<<grid, 256, 0, stream>>>(zn, rowsum, colTilesPerChunk);

  ntx_finalize<<<1, 256, 0, stream>>>(rowsum, selfsim, pos, (float*)d_out, B);
}
